// FullAttention_78503412236707
// MI455X (gfx1250) — matrix operand order repaired and checked by reading, not compiled, not run on hardware
//
#include <hip/hip_runtime.h>

// Problem constants (match reference): B=8, S=2048, E=1024, D=128
#define BB 8
#define SS 2048
#define EE 1024
#define DD 128

typedef __attribute__((ext_vector_type(16))) __bf16 v16bf;
typedef __attribute__((ext_vector_type(4)))  __bf16 v4bf;
typedef __attribute__((ext_vector_type(8)))  float  v8f;

// log2(e) / D : folds softmax base-2 conversion and the reference's 1/D score
// scale (it divides by sqrt(D) twice) into one multiply.
#define SCL (1.4426950408889634f / 128.0f)

// v_exp_f32 is natively exp2
#define EXP2(x) __builtin_amdgcn_exp2f(x)

union Frag16 {
  v16bf bf;
  unsigned int u32[8];
};

// B-fragment: 16 contiguous bf16 (32 bytes) per lane
__device__ __forceinline__ void ldB(Frag16& f, const __bf16* p) {
  const uint4* q = (const uint4*)p;
  uint4 a = q[0], b = q[1];
  f.u32[0]=a.x; f.u32[1]=a.y; f.u32[2]=a.z; f.u32[3]=a.w;
  f.u32[4]=b.x; f.u32[5]=b.y; f.u32[6]=b.z; f.u32[7]=b.w;
}
// A-fragment: two 8-bf16 chunks (16 bytes each) at p and p+8 elements
__device__ __forceinline__ void ldA(Frag16& f, const __bf16* p) {
  uint4 a = *(const uint4*)p;
  uint4 b = *(const uint4*)(p + 8);
  f.u32[0]=a.x; f.u32[1]=a.y; f.u32[2]=a.z; f.u32[3]=a.w;
  f.u32[4]=b.x; f.u32[5]=b.y; f.u32[6]=b.z; f.u32[7]=b.w;
}

// ---------------- Kernel 0: one-time fp32 -> bf16 conversion ----------------
__global__ __launch_bounds__(256) void f32_to_bf16_kernel(
    const float* __restrict__ src, __bf16* __restrict__ dst, int n)
{
  const size_t i = ((size_t)blockIdx.x * blockDim.x + threadIdx.x) * 4;
  if (i + 3 < (size_t)n) {
    float4 f = *(const float4*)(src + i);
    v4bf o = { (__bf16)f.x, (__bf16)f.y, (__bf16)f.z, (__bf16)f.w };
    *(v4bf*)(dst + i) = o;
  }
}

// ---------------- Kernel 1: QKV projection (bf16 in, bf16 out) ----------------
// out[s][d] = sum_e x[s][e] * W[d][e]; one wave per (16-row s-tile, 16-col d-tile, {q,k,v})
// q,k stored bf16 row-major [B][S][D]; v stored transposed [B][D][S] for PV B-fragments.
__global__ __launch_bounds__(32) void qkv_proj_kernel(
    const __bf16* __restrict__ xb, const __bf16* __restrict__ wqb,
    const __bf16* __restrict__ wkb, const __bf16* __restrict__ wvb,
    __bf16* __restrict__ qb, __bf16* __restrict__ kb, __bf16* __restrict__ vt)
{
  const int lane = threadIdx.x & 31;
  const int half = lane >> 4;        // 0 or 1
  const int lm   = lane & 15;
  const int b    = blockIdx.x / (SS / 16);
  const int s0   = (blockIdx.x % (SS / 16)) * 16;
  const int d0   = blockIdx.y * 16;
  const int which = blockIdx.z;
  const __bf16* W = (which == 0) ? wqb : (which == 1) ? wkb : wvb;

  const __bf16* xrow = xb + ((size_t)b * SS + s0 + lm) * EE;  // A: row s0+lm
  const __bf16* wrow = W + (size_t)(d0 + lm) * EE;            // B: column d0+lm

  v8f acc = {};
  for (int e0 = 0; e0 < EE; e0 += 32) {
    if (e0 + 32 < EE) {                         // prefetch next e-block (near cache)
      __builtin_prefetch(xrow + e0 + 32, 0, 3);
      __builtin_prefetch(wrow + e0 + 32, 0, 3);
    }
    Frag16 af, bf;
    ldA(af, xrow + e0 + half * 16);             // K = e0 + 16*half .. +15, contiguous
    ldB(bf, wrow + e0 + half * 16);             // K = e0 + 16*half .. +15, contiguous
    acc = __builtin_amdgcn_wmma_f32_16x16x32_bf16(false, af.bf, false, bf.bf,
                                                  (short)0, acc, false, false);
  }

  // C layout: lane holds column n=lm, rows m = i + 8*half
  if (which == 2) {
    __bf16* dst = vt + ((size_t)b * DD + d0 + lm) * SS + s0 + 8 * half;
    #pragma unroll
    for (int i = 0; i < 8; ++i) dst[i] = (__bf16)acc[i];   // contiguous along s
  } else {
    __bf16* dst = (which == 0 ? qb : kb)
                + ((size_t)b * SS + s0 + 8 * half) * DD + d0 + lm;
    #pragma unroll
    for (int i = 0; i < 8; ++i) dst[(size_t)i * DD] = (__bf16)acc[i];
  }
}

// ---------------- Kernel 2: flash attention ----------------
// One wave per 16-row q-tile; online softmax over 64-key blocks (amortizes the
// shuffle reductions and accumulator rescale over 32 WMMAs); D=128 -> 8 acc tiles.
__global__ __launch_bounds__(128) void flash_attn_kernel(
    const __bf16* __restrict__ qb, const __bf16* __restrict__ kb,
    const __bf16* __restrict__ vt, float* __restrict__ out)
{
  __shared__ __bf16 plds[4][16 * 64];  // per-wave P staging (2 KB each)
  const int lane = threadIdx.x & 31;
  const int wv   = threadIdx.x >> 5;
  const int half = lane >> 4;
  const int lm   = lane & 15;
  const int b    = blockIdx.y;
  const int s0   = (blockIdx.x * 4 + wv) * 16;

  // Q A-fragments for full D=128 (4 chunks of K=32), kept in registers
  Frag16 qf[4];
  {
    const __bf16* qrow = qb + ((size_t)b * SS + s0 + lm) * DD;
    #pragma unroll
    for (int c = 0; c < 4; ++c) ldA(qf[c], qrow + c * 32 + half * 16);
  }

  v8f acc[8] = {};
  float mrow[8], lrow[8];          // running max (log2 domain) and sum
  #pragma unroll
  for (int i = 0; i < 8; ++i) { mrow[i] = -INFINITY; lrow[i] = 0.0f; }

  __bf16* pl = plds[wv];
  const int nblk = (s0 + 16 + 63) >> 6;  // keys 0 .. s0+15, blocks of 64
  for (int blk = 0; blk < nblk; ++blk) {
    const int k0 = blk * 64;
    const __bf16* krow = kb + ((size_t)b * SS + k0 + lm) * DD;

    if (blk + 1 < nblk) {  // prefetch next block's K rows and V segments
      __builtin_prefetch(krow + (size_t)64 * DD + (size_t)(lane - lm) * 2 * DD, 0, 3);
      __builtin_prefetch(krow + (size_t)96 * DD + (size_t)(lane - lm) * 2 * DD, 0, 3);
      __builtin_prefetch(vt + ((size_t)b * DD + lane * 4) * SS + k0 + 64, 0, 3);
    }

    // ---- scores: 16 queries x 64 keys as four 16x16 tiles, f32 acc ----
    v8f ct[4] = {};
    #pragma unroll
    for (int j = 0; j < 4; ++j) {          // key subtile (16 keys each)
      #pragma unroll
      for (int c = 0; c < 4; ++c) {        // D chunk (K=32 each)
        Frag16 kf;
        ldB(kf, krow + (size_t)(j * 16) * DD + c * 32 + half * 16);
        ct[j] = __builtin_amdgcn_wmma_f32_16x16x32_bf16(false, qf[c].bf, false, kf.bf,
                                                        (short)0, ct[j], false, false);
      }
    }

    // ---- scale to log2 domain, mask, online softmax over 64 columns ----
    const bool need_mask = (k0 + 63 > s0);  // only diagonal-crossing block
    float alpha[8];
    #pragma unroll
    for (int i = 0; i < 8; ++i) {
      const int qg = s0 + i + 8 * half;    // global query row
      float v[4];
      #pragma unroll
      for (int j = 0; j < 4; ++j) v[j] = ct[j][i] * SCL;
      if (need_mask) {
        #pragma unroll
        for (int j = 0; j < 4; ++j)
          if (k0 + j * 16 + lm > qg) v[j] = -INFINITY;
      }
      float tm = fmaxf(fmaxf(v[0], v[1]), fmaxf(v[2], v[3]));
      #pragma unroll
      for (int off = 1; off < 16; off <<= 1)
        tm = fmaxf(tm, __shfl_xor(tm, off, 32));
      const float mnew = fmaxf(mrow[i], tm);
      alpha[i] = EXP2(mrow[i] - mnew);
      float p[4], rs = 0.0f;
      #pragma unroll
      for (int j = 0; j < 4; ++j) { p[j] = EXP2(v[j] - mnew); rs += p[j]; }
      #pragma unroll
      for (int off = 1; off < 16; off <<= 1)
        rs += __shfl_xor(rs, off, 32);
      lrow[i] = lrow[i] * alpha[i] + rs;
      mrow[i] = mnew;
      // stage P (bf16) row-major 16x64: row = i+8*half, cols j*16+lm
      const int r = (i + 8 * half) * 64;
      #pragma unroll
      for (int j = 0; j < 4; ++j) pl[r + j * 16 + lm] = (__bf16)p[j];
    }

    // rescale accumulators (once per 64 keys)
    #pragma unroll
    for (int t = 0; t < 8; ++t) {
      #pragma unroll
      for (int i = 0; i < 8; ++i) acc[t][i] *= alpha[i];
    }

    // P back from LDS as two A-fragments (keys k0..+31 and k0+32..+63)
    Frag16 pf0, pf1;
    ldA(pf0, pl + lm * 64 + half * 16);
    ldA(pf1, pl + lm * 64 + 32 + half * 16);

    // ---- P x V: 8 d-tiles x 2 key chunks; V transposed -> contiguous B-frags ----
    #pragma unroll
    for (int t = 0; t < 8; ++t) {
      const __bf16* vrow = vt + ((size_t)b * DD + t * 16 + lm) * SS;
      Frag16 vf;
      ldB(vf, vrow + k0 + half * 16);
      acc[t] = __builtin_amdgcn_wmma_f32_16x16x32_bf16(false, pf0.bf, false, vf.bf,
                                                       (short)0, acc[t], false, false);
      ldB(vf, vrow + k0 + 32 + half * 16);
      acc[t] = __builtin_amdgcn_wmma_f32_16x16x32_bf16(false, pf1.bf, false, vf.bf,
                                                       (short)0, acc[t], false, false);
    }
  }

  // epilogue: z = acc / l, fp32 out [B][S][D]
  #pragma unroll
  for (int t = 0; t < 8; ++t) {
    #pragma unroll
    for (int i = 0; i < 8; ++i) {
      out[((size_t)b * SS + s0 + i + 8 * half) * DD + t * 16 + lm] =
          acc[t][i] / lrow[i];
    }
  }
}

extern "C" void kernel_launch(void* const* d_in, const int* in_sizes, int n_in,
                              void* d_out, int out_size, void* d_ws, size_t ws_size,
                              hipStream_t stream) {
  (void)in_sizes; (void)n_in; (void)out_size; (void)ws_size;
  const float* x  = (const float*)d_in[0];
  const float* Wq = (const float*)d_in[1];
  const float* Wk = (const float*)d_in[2];
  const float* Wv = (const float*)d_in[3];

  // Workspace layout (bf16):
  //   xb  [B][S][E]  32 MB      wq/wk/wv [D][E]  256 KB each
  //   qb,kb [B][S][D] 4 MB each;  vt [B][D][S] 4 MB   (total ~45 MB, L2-resident)
  __bf16* xb  = (__bf16*)d_ws;
  __bf16* wqb = xb  + (size_t)BB * SS * EE;
  __bf16* wkb = wqb + (size_t)DD * EE;
  __bf16* wvb = wkb + (size_t)DD * EE;
  __bf16* qb  = wvb + (size_t)DD * EE;
  __bf16* kb  = qb  + (size_t)BB * SS * DD;
  __bf16* vt  = kb  + (size_t)BB * SS * DD;

  const int nx = BB * SS * EE;           // 16,777,216
  const int nw = DD * EE;                // 131,072
  f32_to_bf16_kernel<<<nx / 4 / 256, 256, 0, stream>>>(x,  xb,  nx);
  f32_to_bf16_kernel<<<nw / 4 / 256, 256, 0, stream>>>(Wq, wqb, nw);
  f32_to_bf16_kernel<<<nw / 4 / 256, 256, 0, stream>>>(Wk, wkb, nw);
  f32_to_bf16_kernel<<<nw / 4 / 256, 256, 0, stream>>>(Wv, wvb, nw);

  dim3 gA(BB * (SS / 16), DD / 16, 3);
  qkv_proj_kernel<<<gA, dim3(32), 0, stream>>>(xb, wqb, wkb, wvb, qb, kb, vt);

  dim3 gB(SS / 64, BB);  // 4 waves/block, one q-tile each
  flash_attn_kernel<<<gB, dim3(128), 0, stream>>>(qb, kb, vt, (float*)d_out);
}
